// Encoder_61237643706772
// MI455X (gfx1250) — compile-verified, matrix operand order checked by reading, not run on hardware
//
#include <hip/hip_runtime.h>
#include <hip/hip_bf16.h>

typedef __attribute__((ext_vector_type(16))) __bf16 v16bf;
typedef __attribute__((ext_vector_type(8)))  __bf16 v8bf;
typedef __attribute__((ext_vector_type(8)))  float  v8f;

#define BATCH 128
#define SEQ   1024
#define HID1  128
#define HID2  64
#define G1    (4*HID1)   // 512 gate outputs, layer 1
#define G2    (4*HID2)   // 256 gate outputs, layer 2
#define ROWS  16         // batch rows per workgroup (WMMA M)

// LDS layout (bytes):
//  bf16 region: Whh1 512x128, Wih2 256x128, Whh2 256x64, H1 16x128, H2 16x64
//  f32  region: B1 512, Wi1 512, B2 256, C1 16x128, C2 16x64, Gt1 16x512, Gt2 16x256
// total = 235,520 + 66,560 = 302,080 B  (< 320 KB per WGP)
#define LDS_BYTES 302080

__device__ __forceinline__ float sigm(float x) {
    return 1.0f / (1.0f + __expf(-x));
}
__device__ __forceinline__ float tanh_fast(float x) {
    return 1.0f - 2.0f / (__expf(2.0f * x) + 1.0f);
}

// Load one 16x32 bf16 operand tile per the CDNA5 WMMA register layout:
// lane (with laneSel = lane>>4) holds K = [k0+sel*8, +8) in VGPR0..3 and
// K = [k0+16+sel*8, +8) in VGPR4..7 for its row.  Row-major LDS, 16B-aligned.
__device__ __forceinline__ v16bf load_tile16(const __bf16* __restrict__ base,
                                             int row, int ldk, int k0, int laneSel) {
    const __bf16* p = base + row * ldk + k0 + laneSel * 8;
    union { v16bf v; v8bf h[2]; } u;
    u.h[0] = *(const v8bf*)(p);
    u.h[1] = *(const v8bf*)(p + 16);
    return u.v;
}

extern "C" __global__ __launch_bounds__(256, 1)
void lstm2_fused(const float* __restrict__ x,
                 const float* __restrict__ w_ih1, const float* __restrict__ w_hh1,
                 const float* __restrict__ b_ih1, const float* __restrict__ b_hh1,
                 const float* __restrict__ w_ih2, const float* __restrict__ w_hh2,
                 const float* __restrict__ b_ih2, const float* __restrict__ b_hh2,
                 float* __restrict__ out)
{
    extern __shared__ __align__(16) unsigned char smem[];
    __bf16* sWhh1 = (__bf16*)smem;              // 512*128
    __bf16* sWih2 = sWhh1 + G1 * HID1;          // 256*128
    __bf16* sWhh2 = sWih2 + G2 * HID1;          // 256*64
    __bf16* sH1   = sWhh2 + G2 * HID2;          // 16*128
    __bf16* sH2   = sH1 + ROWS * HID1;          // 16*64
    float* sB1  = (float*)(sH2 + ROWS * HID2);  // 512  (b_ih1+b_hh1)
    float* sWi1 = sB1 + G1;                     // 512  (w_ih1 column, F==1)
    float* sB2  = sWi1 + G1;                    // 256  (b_ih2+b_hh2)
    float* sC1  = sB2 + G2;                     // 16*128
    float* sC2  = sC1 + ROWS * HID1;            // 16*64
    float* sG1  = sC2 + ROWS * HID2;            // 16*512 gate staging L1
    float* sG2  = sG1 + ROWS * G1;              // 16*256 gate staging L2

    const int tid     = threadIdx.x;
    const int wave    = tid >> 5;
    const int lane    = tid & 31;
    const int laneLo  = lane & 15;
    const int laneSel = lane >> 4;
    const int bBase   = blockIdx.x * ROWS;

    // ---- one-time setup: fp32 -> bf16 weights into LDS, fold biases, zero state
    for (int i = tid; i < G1; i += 256) { sB1[i] = b_ih1[i] + b_hh1[i]; sWi1[i] = w_ih1[i]; }
    for (int i = tid; i < G2; i += 256) { sB2[i] = b_ih2[i] + b_hh2[i]; }
    for (int i = tid; i < G1 * HID1; i += 256) sWhh1[i] = (__bf16)w_hh1[i];
    for (int i = tid; i < G2 * HID1; i += 256) sWih2[i] = (__bf16)w_ih2[i];
    for (int i = tid; i < G2 * HID2; i += 256) sWhh2[i] = (__bf16)w_hh2[i];
    for (int i = tid; i < ROWS * HID1; i += 256) { sC1[i] = 0.0f; sH1[i] = (__bf16)0.0f; }
    for (int i = tid; i < ROWS * HID2; i += 256) { sC2[i] = 0.0f; sH2[i] = (__bf16)0.0f; }
    __syncthreads();

    const int m  = tid >> 4;          // 0..15: batch row for elementwise phases
    const int jb = tid & 15;

#pragma unroll 1
    for (int t = 0; t < SEQ; ++t) {
        // ===== Layer 1 recurrent GEMM: sG1 = H1 (16x128) @ Whh1^T (128x512) =====
        v16bf a1[4];
#pragma unroll
        for (int kk = 0; kk < 4; ++kk)
            a1[kk] = load_tile16(sH1, laneLo, HID1, kk * 32, laneSel);
#pragma unroll
        for (int tt = 0; tt < 4; ++tt) {
            const int n0 = (wave * 4 + tt) * 16;
            v8f acc = {0.f, 0.f, 0.f, 0.f, 0.f, 0.f, 0.f, 0.f};
#pragma unroll
            for (int kk = 0; kk < 4; ++kk) {
                v16bf b = load_tile16(sWhh1, n0 + laneLo, HID1, kk * 32, laneSel);
                acc = __builtin_amdgcn_wmma_f32_16x16x32_bf16(
                        false, a1[kk], false, b, (short)0, acc, false, false);
            }
            const int n = n0 + laneLo;
#pragma unroll
            for (int r = 0; r < 8; ++r)
                sG1[(laneSel * 8 + r) * G1 + n] = acc[r];
        }
        __syncthreads();

        // ===== Layer 1 gate update (i,f,g,o), write h1 as bf16 for next GEMM =====
        {
            const int j0 = jb * 8;
            const float xv = x[(bBase + m) * SEQ + t];   // F==1: input contribution is xv*w_ih1
            const float* g = sG1 + m * G1;
#pragma unroll
            for (int jj = 0; jj < 8; ++jj) {
                const int j = j0 + jj;
                float pi = g[j]            + sB1[j]            + xv * sWi1[j];
                float pf = g[HID1 + j]     + sB1[HID1 + j]     + xv * sWi1[HID1 + j];
                float pg = g[2 * HID1 + j] + sB1[2 * HID1 + j] + xv * sWi1[2 * HID1 + j];
                float po = g[3 * HID1 + j] + sB1[3 * HID1 + j] + xv * sWi1[3 * HID1 + j];
                float c  = sigm(pf) * sC1[m * HID1 + j] + sigm(pi) * tanh_fast(pg);
                float h  = sigm(po) * tanh_fast(c);
                sC1[m * HID1 + j] = c;
                sH1[m * HID1 + j] = (__bf16)h;
            }
        }
        __syncthreads();

        // ===== Layer 2 GEMMs: sG2 = H1 @ Wih2^T + H2 @ Whh2^T =====
        v16bf ah1[4];
#pragma unroll
        for (int kk = 0; kk < 4; ++kk)
            ah1[kk] = load_tile16(sH1, laneLo, HID1, kk * 32, laneSel);
        v16bf ah2[2];
#pragma unroll
        for (int kk = 0; kk < 2; ++kk)
            ah2[kk] = load_tile16(sH2, laneLo, HID2, kk * 32, laneSel);
#pragma unroll
        for (int tt = 0; tt < 2; ++tt) {
            const int n0 = (wave * 2 + tt) * 16;
            v8f acc = {0.f, 0.f, 0.f, 0.f, 0.f, 0.f, 0.f, 0.f};
#pragma unroll
            for (int kk = 0; kk < 4; ++kk) {
                v16bf b = load_tile16(sWih2, n0 + laneLo, HID1, kk * 32, laneSel);
                acc = __builtin_amdgcn_wmma_f32_16x16x32_bf16(
                        false, ah1[kk], false, b, (short)0, acc, false, false);
            }
#pragma unroll
            for (int kk = 0; kk < 2; ++kk) {
                v16bf b = load_tile16(sWhh2, n0 + laneLo, HID2, kk * 32, laneSel);
                acc = __builtin_amdgcn_wmma_f32_16x16x32_bf16(
                        false, ah2[kk], false, b, (short)0, acc, false, false);
            }
            const int n = n0 + laneLo;
#pragma unroll
            for (int r = 0; r < 8; ++r)
                sG2[(laneSel * 8 + r) * G2 + n] = acc[r];
        }
        __syncthreads();

        // ===== Layer 2 gate update; final step streams h2 to global output =====
        {
            const int j0 = jb * 4;
            const float* g = sG2 + m * G2;
#pragma unroll
            for (int jj = 0; jj < 4; ++jj) {
                const int j = j0 + jj;
                float pi = g[j]            + sB2[j];
                float pf = g[HID2 + j]     + sB2[HID2 + j];
                float pg = g[2 * HID2 + j] + sB2[2 * HID2 + j];
                float po = g[3 * HID2 + j] + sB2[3 * HID2 + j];
                float c  = sigm(pf) * sC2[m * HID2 + j] + sigm(pi) * tanh_fast(pg);
                float h  = sigm(po) * tanh_fast(c);
                sC2[m * HID2 + j] = c;
                sH2[m * HID2 + j] = (__bf16)h;
                if (t == SEQ - 1)
                    out[(bBase + m) * HID2 + j] = h;   // (B, 1, EMB) flat
            }
        }
        __syncthreads();
    }
}

extern "C" void kernel_launch(void* const* d_in, const int* in_sizes, int n_in,
                              void* d_out, int out_size, void* d_ws, size_t ws_size,
                              hipStream_t stream) {
    (void)in_sizes; (void)n_in; (void)out_size; (void)d_ws; (void)ws_size;
    const float* x     = (const float*)d_in[0];
    const float* w_ih1 = (const float*)d_in[1];
    const float* w_hh1 = (const float*)d_in[2];
    const float* b_ih1 = (const float*)d_in[3];
    const float* b_hh1 = (const float*)d_in[4];
    const float* w_ih2 = (const float*)d_in[5];
    const float* w_hh2 = (const float*)d_in[6];
    const float* b_ih2 = (const float*)d_in[7];
    const float* b_hh2 = (const float*)d_in[8];
    float* out = (float*)d_out;

    // Allow > default dynamic LDS (benign if repeated; not a stream op).
    (void)hipFuncSetAttribute(reinterpret_cast<const void*>(lstm2_fused),
                              hipFuncAttributeMaxDynamicSharedMemorySize,
                              (int)LDS_BYTES);

    lstm2_fused<<<BATCH / ROWS, 256, LDS_BYTES, stream>>>(
        x, w_ih1, w_hh1, b_ih1, b_hh1, w_ih2, w_hh2, b_ih2, b_hh2, out);
}
